// MixedNet_61753039781957
// MI455X (gfx1250) — matrix operand order repaired and checked by reading, not compiled, not run on hardware
//
#include <hip/hip_runtime.h>
#include <hip/hip_bf16.h>

// MixedNet forward for MI455X (gfx1250, wave32, WMMA).
// Pipeline:
//   1) one-time kernels: split x -> bf16(hi,lo); transpose+split each W -> [N][K] bf16(hi,lo)
//   2) GEMM: C = A@B via split-precision bf16x2 (hi*hi + hi*lo + lo*hi) with
//      v_wmma_f32_16x16x32_bf16; epilogue fuses sin/tanh/log + alpha and writes the
//      next layer's A already split into bf16(hi,lo). Last layer writes fp32.
// Inner GEMM loop is pure b128 global load -> b128 LDS store -> b128 LDS load -> WMMA,
// with register-level prefetch of the next K-tile overlapping the WMMA block.
// Workspace: splits + transposed weights + 2 split ping-pong h buffers = 304 MB.

#define BM 128
#define BN 128
#define BK 32
#define KPAD 40  // BK + 8 bf16 elems padding: 80B row stride, 16B aligned, no bank conflicts

typedef __attribute__((ext_vector_type(16))) __bf16 v16bf;
typedef __attribute__((ext_vector_type(8)))  float  v8f;
typedef unsigned short u16;

union FragU { uint4 q[2]; v16bf v; };

__device__ __forceinline__ u16 f32_bf16_rne(float f) {
  unsigned u = __float_as_uint(f);
  return (u16)((u + 0x7FFFu + ((u >> 16) & 1u)) >> 16);
}
__device__ __forceinline__ float bf16_f32(u16 h) {
  return __uint_as_float(((unsigned)h) << 16);
}

// ---------------- one-time conversion kernels ----------------

// elementwise split: hi = bf16(x), lo = bf16(x - hi)
__global__ __launch_bounds__(256) void split_kernel(const float* __restrict__ src,
                                                    u16* __restrict__ hi,
                                                    u16* __restrict__ lo, int n4) {
  int i = blockIdx.x * blockDim.x + threadIdx.x;
  if (i >= n4) return;
  float4 v = *(const float4*)(src + (size_t)i * 4);
  const float f[4] = {v.x, v.y, v.z, v.w};
  unsigned hp[2], lp[2];
#pragma unroll
  for (int j = 0; j < 2; ++j) {
    u16 h0 = f32_bf16_rne(f[2 * j]), h1 = f32_bf16_rne(f[2 * j + 1]);
    u16 l0 = f32_bf16_rne(f[2 * j] - bf16_f32(h0));
    u16 l1 = f32_bf16_rne(f[2 * j + 1] - bf16_f32(h1));
    hp[j] = (unsigned)h0 | ((unsigned)h1 << 16);
    lp[j] = (unsigned)l0 | ((unsigned)l1 << 16);
  }
  *(uint2*)(hi + (size_t)i * 4) = make_uint2(hp[0], hp[1]);
  *(uint2*)(lo + (size_t)i * 4) = make_uint2(lp[0], lp[1]);
}

// W[K][N] fp32 -> Thi/Tlo[N][K] bf16 (32x32 LDS tiles)
__global__ __launch_bounds__(256) void transpose_split_kernel(
    const float* __restrict__ W, u16* __restrict__ Thi, u16* __restrict__ Tlo,
    int K, int N) {
  __shared__ float tile[32][33];
  const int n0 = blockIdx.x * 32, k0 = blockIdx.y * 32;
  const int tx = threadIdx.x & 31, ty = threadIdx.x >> 5;  // 32 x 8
#pragma unroll
  for (int i = 0; i < 4; ++i) {
    int r = ty + i * 8;
    tile[r][tx] = W[(size_t)(k0 + r) * N + n0 + tx];
  }
  __syncthreads();
#pragma unroll
  for (int i = 0; i < 4; ++i) {
    int r = ty + i * 8;
    float f = tile[tx][r];  // = W[k0+tx][n0+r]
    u16 h = f32_bf16_rne(f);
    Thi[(size_t)(n0 + r) * K + k0 + tx] = h;
    Tlo[(size_t)(n0 + r) * K + k0 + tx] = f32_bf16_rne(f - bf16_f32(h));
  }
}

// ---------------- WMMA fragment loads from LDS ----------------

__device__ __forceinline__ v16bf load_afrag(const u16* As, int mbase, int lane) {
  const int half = lane >> 4;               // lanes 0-15: K 0..7/16..23 ; 16-31: +8
  const int row  = mbase + (lane & 15);
  const u16* p = As + row * KPAD + half * 8;
  FragU f;
  f.q[0] = *(const uint4*)(p);
  f.q[1] = *(const uint4*)(p + 16);
  return f.v;
}
__device__ __forceinline__ v16bf load_bfrag(const u16* Bs, int nbase, int lane) {
  const int half = lane >> 4;               // lanes 0-15: K 0..7/16..23 ; 16-31: +8
  const int col  = nbase + (lane & 15);
  const u16* p = Bs + col * KPAD + half * 8;
  FragU f;
  f.q[0] = *(const uint4*)(p);
  f.q[1] = *(const uint4*)(p + 16);
  return f.v;
}

// ---------------- split-precision GEMM ----------------
// A (hi/lo): [M][K] bf16 row-major.  B (hi/lo): [N][K] bf16 (pre-transposed).
// mode 0: y = act(x)*alpha per column segment, write split bf16 to Chi/Clo.
// mode 1: write fp32 to Cf32.
struct Stage { uint4 r[8]; };

__global__ __launch_bounds__(256) void gemm_split_bf16(
    const u16* __restrict__ Ahi, const u16* __restrict__ Alo,
    const u16* __restrict__ Bhi, const u16* __restrict__ Blo,
    u16* __restrict__ Chi, u16* __restrict__ Clo, float* __restrict__ Cf32,
    int M, int N, int K, const float* __restrict__ alpha, int mode) {

  __shared__ __align__(16) u16 As_hi[BM * KPAD];
  __shared__ __align__(16) u16 As_lo[BM * KPAD];
  __shared__ __align__(16) u16 Bs_hi[BN * KPAD];
  __shared__ __align__(16) u16 Bs_lo[BN * KPAD];

  const int tid   = threadIdx.x;
  const int lane  = tid & 31;
  const int wave  = tid >> 5;
  const int waveM = wave >> 2;  // 0..1 -> 64 rows
  const int waveN = wave & 3;   // 0..3 -> 32 cols
  const int m0 = blockIdx.y * BM;
  const int n0 = blockIdx.x * BN;

  v8f acc[4][2] = {};
  Stage st;

  auto load_stage = [&](int k0) {
#pragma unroll
    for (int i = 0; i < 2; ++i) {
      int idx = tid + i * 256;
      int row = idx >> 2;           // 0..127
      int c8  = (idx & 3) * 8;      // 0,8,16,24
      size_t offA = (size_t)(m0 + row) * K + k0 + c8;
      size_t offB = (size_t)(n0 + row) * K + k0 + c8;
      st.r[i]     = *(const uint4*)(Ahi + offA);
      st.r[2 + i] = *(const uint4*)(Alo + offA);
      st.r[4 + i] = *(const uint4*)(Bhi + offB);
      st.r[6 + i] = *(const uint4*)(Blo + offB);
    }
  };
  auto store_stage = [&]() {
#pragma unroll
    for (int i = 0; i < 2; ++i) {
      int idx = tid + i * 256;
      int row = idx >> 2;
      int c8  = (idx & 3) * 8;
      int off = row * KPAD + c8;
      *(uint4*)(As_hi + off) = st.r[i];
      *(uint4*)(As_lo + off) = st.r[2 + i];
      *(uint4*)(Bs_hi + off) = st.r[4 + i];
      *(uint4*)(Bs_lo + off) = st.r[6 + i];
    }
  };

  load_stage(0);
  store_stage();

  for (int k0 = 0;;) {
    __syncthreads();                       // LDS tile ready for all waves
    const int kn = k0 + BK;
    const bool more = kn < K;
    if (more) load_stage(kn);              // global loads in flight during WMMAs

    // 3 passes: hi*hi, hi*lo, lo*hi
#pragma unroll
    for (int pass = 0; pass < 3; ++pass) {
      const u16* Asel = (pass == 2) ? As_lo : As_hi;
      const u16* Bsel = (pass == 1) ? Bs_lo : Bs_hi;
      v16bf af[4];
#pragma unroll
      for (int mt = 0; mt < 4; ++mt)
        af[mt] = load_afrag(Asel, waveM * 64 + mt * 16, lane);
      v16bf bfm[2];
#pragma unroll
      for (int nt = 0; nt < 2; ++nt)
        bfm[nt] = load_bfrag(Bsel, waveN * 32 + nt * 16, lane);
#pragma unroll
      for (int mt = 0; mt < 4; ++mt)
#pragma unroll
        for (int nt = 0; nt < 2; ++nt)
          acc[mt][nt] = __builtin_amdgcn_wmma_f32_16x16x32_bf16(
              false, af[mt], false, bfm[nt], (short)0, acc[mt][nt], false, false);
    }

    if (!more) break;
    __syncthreads();                       // all reads of tile done
    store_stage();                         // write prefetched tile
    k0 = kn;
  }

  // epilogue: lane<16 -> M=base+r, lane>=16 -> M=base+8+r ; N = base+(lane&15)
  const int colb = n0 + waveN * 32 + (lane & 15);
  const int rowb = m0 + waveM * 64 + ((lane >> 4) << 3);
#pragma unroll
  for (int nt = 0; nt < 2; ++nt) {
    const int col = colb + nt * 16;
    const float av = (mode == 0) ? alpha[col] : 1.0f;
#pragma unroll
    for (int mt = 0; mt < 4; ++mt) {
      v8f c = acc[mt][nt];
#pragma unroll
      for (int r = 0; r < 8; ++r) {
        const int row = rowb + mt * 16 + r;
        float x = c[r];
        if (mode == 0) {
          float y;
          if (col < 2048)       y = __sinf(x);
          else if (col < 3072)  y = tanhf(x);
          else                  y = __logf(x * x);
          y *= av;
          u16 h = f32_bf16_rne(y);
          Chi[(size_t)row * N + col] = h;
          Clo[(size_t)row * N + col] = f32_bf16_rne(y - bf16_f32(h));
        } else {
          Cf32[(size_t)row * N + col] = x;
        }
      }
    }
  }
}

// ---------------- host ----------------

extern "C" void kernel_launch(void* const* d_in, const int* in_sizes, int n_in,
                              void* d_out, int out_size, void* d_ws, size_t ws_size,
                              hipStream_t stream) {
  (void)in_sizes; (void)n_in; (void)out_size; (void)ws_size;
  const float* x  = (const float*)d_in[0];
  const float* W0 = (const float*)d_in[1];
  const float* W1 = (const float*)d_in[2];
  const float* W2 = (const float*)d_in[3];
  const float* W3 = (const float*)d_in[4];
  const float* a0 = (const float*)d_in[5];
  const float* a1 = (const float*)d_in[6];
  const float* a2 = (const float*)d_in[7];
  float* out = (float*)d_out;

  const int B = 4096, Din = 1024, H = 4096, Dout = 1024;
  const size_t SZ_XW0 = (size_t)B * Din;   // 4M
  const size_t SZ_H   = (size_t)B * H;     // 16M
  const size_t SZ_W3  = (size_t)Dout * H;  // 4M

  u16* p = (u16*)d_ws;
  u16* Xhi = p; p += SZ_XW0;  u16* Xlo = p; p += SZ_XW0;
  u16* W0h = p; p += SZ_XW0;  u16* W0l = p; p += SZ_XW0;   // [H][Din]
  u16* W1h = p; p += SZ_H;    u16* W1l = p; p += SZ_H;     // [H][H]
  u16* W2h = p; p += SZ_H;    u16* W2l = p; p += SZ_H;     // [H][H]
  u16* W3h = p; p += SZ_W3;   u16* W3l = p; p += SZ_W3;    // [Dout][H]
  u16* H0h = p; p += SZ_H;    u16* H0l = p; p += SZ_H;
  u16* H1h = p; p += SZ_H;    u16* H1l = p; p += SZ_H;

  dim3 blk(256);
  // one-time conversions
  split_kernel<<<dim3((unsigned)(SZ_XW0 / 4 / 256)), blk, 0, stream>>>(x, Xhi, Xlo, (int)(SZ_XW0 / 4));
  transpose_split_kernel<<<dim3(H / 32, Din / 32), blk, 0, stream>>>(W0, W0h, W0l, Din, H);
  transpose_split_kernel<<<dim3(H / 32, H / 32),   blk, 0, stream>>>(W1, W1h, W1l, H, H);
  transpose_split_kernel<<<dim3(H / 32, H / 32),   blk, 0, stream>>>(W2, W2h, W2l, H, H);
  transpose_split_kernel<<<dim3(Dout / 32, H / 32), blk, 0, stream>>>(W3, W3h, W3l, H, Dout);

  dim3 gH(H / BN, B / BM);
  dim3 gO(Dout / BN, B / BM);
  gemm_split_bf16<<<gH, blk, 0, stream>>>(Xhi, Xlo, W0h, W0l, H0h, H0l, nullptr, B, H, Din, a0, 0);
  gemm_split_bf16<<<gH, blk, 0, stream>>>(H0h, H0l, W1h, W1l, H1h, H1l, nullptr, B, H, H, a1, 0);
  gemm_split_bf16<<<gH, blk, 0, stream>>>(H1h, H1l, W2h, W2l, H0h, H0l, nullptr, B, H, H, a2, 0);
  gemm_split_bf16<<<gO, blk, 0, stream>>>(H0h, H0l, W3h, W3l, nullptr, nullptr, out, B, Dout, H, nullptr, 1);
}
